// LunchboxMHSA_17008070492871
// MI455X (gfx1250) — matrix operand order repaired and checked by reading, not compiled, not run on hardware
//
#include <hip/hip_runtime.h>
#include <hip/hip_bf16.h>
#include <stddef.h>

// LunchboxMHSA, MI455X (gfx1250, wave32, WMMA).
// Algebraic collapse: attention == per-channel gain g[b,c]; real work is two
// bf16 WMMA GEMMs (x@Wv, then regrouped @ proj_w), fused through LDS.

typedef __bf16 bf16_t;
typedef __attribute__((ext_vector_type(16))) __bf16 v16bf;
typedef __attribute__((ext_vector_type(8)))  __bf16 v8bf;
typedef __attribute__((ext_vector_type(8)))  float  v8f;

namespace {
constexpr int   kB  = 8;
constexpr int   kN  = 4096;
constexpr int   kD  = 256;
constexpr int   kH  = 8;
constexpr int   kHD = 2048;          // H*D
constexpr float kScale = 0.0625f;    // max(D,R)^-0.5 = 1/16
constexpr int   kXT = 264;           // padded LDS row stride (bf16 elems), 16B-aligned rows
constexpr int   kS  = 16;            // splits for x row-sum reduction
}

// ---- Pass 1a: partial row sums of x over n ------------------------------
__global__ void k_xs1(const float* __restrict__ x, float* __restrict__ part) {
  const int blk = blockIdx.x;
  const int b = blk / kS, s = blk % kS;
  const int k = threadIdx.x;
  const float* p = x + ((size_t)b * kN + (size_t)s * (kN / kS)) * kD + k;
  float acc = 0.f;
  for (int n = 0; n < kN / kS; ++n) acc += p[(size_t)n * kD];
  part[((size_t)b * kS + s) * kD + k] = acc;
}

// ---- Pass 1b: finish Xs[b,k] = sum_n x[b,n,k] ---------------------------
__global__ void k_xs2(const float* __restrict__ part, float* __restrict__ Xs) {
  const int b = blockIdx.x, k = threadIdx.x;
  float acc = 0.f;
  for (int s = 0; s < kS; ++s) acc += part[((size_t)b * kS + s) * kD + k];
  Xs[b * kD + k] = acc;
}

// ---- Pass 2: gains g[b,c] = scale * Sq[b,c] * L[c%256]; gb = g * bias_v --
__global__ void k_g(const float* __restrict__ Xs, const float* __restrict__ qkv_w,
                    const float* __restrict__ qkv_b, const float* __restrict__ lunchbox,
                    float* __restrict__ g, float* __restrict__ gb) {
  const int idx = blockIdx.x * 256 + threadIdx.x;   // 0 .. 8*2048-1
  const int b = idx >> 11, c = idx & 2047;
  float dot = 0.f;
  for (int k = 0; k < kD; ++k) dot += Xs[b * kD + k] * qkv_w[(size_t)k * (2 * kHD) + c];
  const float Sq = dot + (float)kN * qkv_b[c];
  const int d = c & 255;
  float L = 0.f;
  for (int r = 0; r < 32; ++r) L += lunchbox[d * 32 + r];
  const float gg = kScale * Sq * L;
  g[(size_t)b * kHD + c]  = gg;
  gb[(size_t)b * kHD + c] = gg * qkv_b[kHD + c];   // gain applied to v-bias
}

// ---- Pass 3: transpose+convert weights to bf16 fragment-friendly layout --
// WvT[c][k] = qkv_w[k][2048+c]  (2048 x 256); PwT[c][j] = proj_w[j][c] (256 x 2048)
__global__ void k_w(const float* __restrict__ qkv_w, const float* __restrict__ proj_w,
                    bf16_t* __restrict__ WvT, bf16_t* __restrict__ PwT) {
  const int idx = blockIdx.x * 256 + threadIdx.x;  // 0 .. 524287
  const int c = idx >> 8, k = idx & 255;
  WvT[idx] = (bf16_t)qkv_w[(size_t)k * (2 * kHD) + kHD + c];
  const int c2 = idx >> 11, j = idx & 2047;
  PwT[idx] = (bf16_t)proj_w[(size_t)j * kD + c2];
}

// ---- Main fused kernel: per WG = (b, 128 seq rows), 8 waves --------------
__global__ __launch_bounds__(256) void lb_main(
    const float* __restrict__ x, const bf16_t* __restrict__ WvT,
    const bf16_t* __restrict__ PwT, const float* __restrict__ g,
    const float* __restrict__ gb, const float* __restrict__ proj_b,
    float* __restrict__ out) {
  extern __shared__ char smem[];
  bf16_t* sbuf = (bf16_t*)smem;                          // 128*kXT: xt, then reused as vh
  float*  s_g  = (float*)(smem + (size_t)128 * kXT * 2); // 256
  float*  s_gb = s_g + 256;
  float*  s_pb = s_gb + 256;

  const int b = blockIdx.x >> 5;
  const int chunk = blockIdx.x & 31;
  const int n0 = chunk * 128;
  const int tid = threadIdx.x;
  const int wave = tid >> 5;
  const int lane = tid & 31;
  const int lr = lane & 15;   // fragment row/col within 16
  const int lh = lane >> 4;   // half select

  // Stage x tile (128 x 256 f32 -> bf16 LDS)
  const float* xb = x + ((size_t)b * kN + n0) * kD;
  for (int idx = tid; idx < 128 * 64; idx += 256) {
    const int r = idx >> 6, q = idx & 63;
    const float4 v = ((const float4*)(xb + (size_t)r * kD))[q];
    bf16_t* dst = &sbuf[r * kXT + q * 4];
    dst[0] = (bf16_t)v.x; dst[1] = (bf16_t)v.y;
    dst[2] = (bf16_t)v.z; dst[3] = (bf16_t)v.w;
  }
  s_pb[tid] = proj_b[tid];
  __syncthreads();

  // Preload this wave's GEMM1 A fragments (16 rows x K=256) into registers.
  // A frag, this code's contiguous K order (B matches): lane%16 = row M; elems 0..15 <-> K=k0+lh*16+e
  v16bf afrag[8];
  {
    const int row = wave * 16 + lr;
    #pragma unroll
    for (int ks = 0; ks < 8; ++ks) {
      const int k0 = ks * 32;
      const v8bf lo = *(const v8bf*)&sbuf[row * kXT + k0 + lh * 16];
      const v8bf hi = *(const v8bf*)&sbuf[row * kXT + k0 + lh * 16 + 8];
      v16bf a;
      #pragma unroll
      for (int e = 0; e < 8; ++e) { a[e] = lo[e]; a[e + 8] = hi[e]; }
      afrag[ks] = a;
    }
  }

  for (int h = 0; h < kH; ++h) {
    s_g[tid]  = g[(size_t)b * kHD + h * kD + tid];
    s_gb[tid] = gb[(size_t)b * kHD + h * kD + tid];
    __syncthreads();  // orders: afrag/xt reads & prev-head vh reads BEFORE new vh writes

    // GEMM1: vh[128x256] = xt @ Wv_h, scaled by gain, stored bf16 in LDS.
    // Wave handles M-band = wave; loops 16 N-tiles; K = 8 x 32.
    #pragma unroll 1
    for (int nt = 0; nt < 16; ++nt) {
      v8f acc = {0.f, 0.f, 0.f, 0.f, 0.f, 0.f, 0.f, 0.f};
      const int c = h * kD + nt * 16 + lr;              // B col = lane%16
      const bf16_t* wp = WvT + (size_t)c * kD + lh * 16;
      #pragma unroll
      for (int ks = 0; ks < 8; ++ks) {
        const v16bf bfrag = *(const v16bf*)(wp + ks * 32);  // K = k0+lh*16+e contiguous
        acc = __builtin_amdgcn_wmma_f32_16x16x32_bf16(
            false, afrag[ks], false, bfrag, (short)0, acc, false, false);
      }
      const float gv  = s_g[nt * 16 + lr];
      const float gbv = s_gb[nt * 16 + lr];
      #pragma unroll
      for (int p = 0; p < 8; ++p) {                     // C/D: row = lh*8+p, col = lr
        const int row = wave * 16 + lh * 8 + p;
        sbuf[row * kXT + nt * 16 + lr] = (bf16_t)(acc[p] * gv + gbv);
      }
    }
    __syncthreads();

    // GEMM2: out[16 x 256] for rows i = h*512 + chunk*16 + rr, K = 2048 over
    // regrouped vh: A2[rr][j] = vh[rr*8 + j/256][j%256]. Wave handles 2 N-tiles.
    #pragma unroll 1
    for (int t = 0; t < 2; ++t) {
      const int nt = wave * 2 + t;
      const int c = nt * 16 + lr;
      v8f acc = {0.f, 0.f, 0.f, 0.f, 0.f, 0.f, 0.f, 0.f};
      const bf16_t* pp = PwT + (size_t)c * kHD + lh * 16;
      #pragma unroll 4
      for (int ks = 0; ks < 64; ++ks) {
        const int k0 = ks * 32;
        const int nloc = lr * 8 + (k0 >> 8);            // 32-chunk never crosses 256-block
        const int dbase = k0 & 255;
        const v8bf lo = *(const v8bf*)&sbuf[nloc * kXT + dbase + lh * 16];
        const v8bf hi = *(const v8bf*)&sbuf[nloc * kXT + dbase + lh * 16 + 8];
        v16bf a2;
        #pragma unroll
        for (int e = 0; e < 8; ++e) { a2[e] = lo[e]; a2[e + 8] = hi[e]; }
        const v16bf bfrag = *(const v16bf*)(pp + k0);
        acc = __builtin_amdgcn_wmma_f32_16x16x32_bf16(
            false, a2, false, bfrag, (short)0, acc, false, false);
      }
      const int ibase = h * 512 + chunk * 16;
      const float pbv = s_pb[nt * 16 + lr];
      float* op = out + ((size_t)b * kN + ibase) * kD + nt * 16 + lr;
      #pragma unroll
      for (int p = 0; p < 8; ++p)
        op[(size_t)(lh * 8 + p) * kD] = acc[p] + pbv;
    }
  }
}

extern "C" void kernel_launch(void* const* d_in, const int* in_sizes, int n_in,
                              void* d_out, int out_size, void* d_ws, size_t ws_size,
                              hipStream_t stream) {
  const float* x        = (const float*)d_in[0];
  const float* qkv_w    = (const float*)d_in[1];
  const float* qkv_b    = (const float*)d_in[2];
  const float* lunchbox = (const float*)d_in[3];
  const float* proj_w   = (const float*)d_in[4];
  const float* proj_b   = (const float*)d_in[5];
  float* out = (float*)d_out;

  // Workspace layout (bytes): Xs 8K | g 64K | gb 64K | WvT 1M | PwT 1M | part 128K
  char* ws = (char*)d_ws;
  float*  Xs   = (float*)(ws);
  float*  g    = (float*)(ws + 8192);
  float*  gb   = (float*)(ws + 8192 + 65536);
  bf16_t* WvT  = (bf16_t*)(ws + 8192 + 131072);
  bf16_t* PwT  = (bf16_t*)(ws + 8192 + 131072 + 1048576);
  float*  part = (float*)(ws + 8192 + 131072 + 2097152);

  k_xs1<<<dim3(kB * kS), dim3(256), 0, stream>>>(x, part);
  k_xs2<<<dim3(kB), dim3(256), 0, stream>>>(part, Xs);
  k_g  <<<dim3(64), dim3(256), 0, stream>>>(Xs, qkv_w, qkv_b, lunchbox, g, gb);
  k_w  <<<dim3(2048), dim3(256), 0, stream>>>(qkv_w, proj_w, WvT, PwT);

  const size_t shmem = (size_t)128 * kXT * 2 + 3 * 256 * sizeof(float);  // ~70.7 KB
  (void)hipFuncSetAttribute((const void*)lb_main,
                            hipFuncAttributeMaxDynamicSharedMemorySize, (int)shmem);
  lb_main<<<dim3(kB * 32), dim3(256), shmem, stream>>>(x, WvT, PwT, g, gb, proj_b, out);
}
